// VSSBlockBS_9371618640305
// MI455X (gfx1250) — matrix operand order repaired and checked by reading, not compiled, not run on hardware
//
#include <hip/hip_runtime.h>
#include <hip/hip_bf16.h>
#include <math.h>

typedef _Float16 f16;
typedef __attribute__((ext_vector_type(16))) _Float16 v16h;
typedef __attribute__((ext_vector_type(8)))  _Float16 v8h;
typedef __attribute__((ext_vector_type(8)))  float    v8f;

// ---- problem constants ----
constexpr int Bn  = 8;
constexpr int Hh  = 32;
constexpr int Ww  = 32;
constexpr int Cc  = 192;
constexpr int DI  = 384;   // 2*C
constexpr int S   = 16;
constexpr int Rr  = 12;
constexpr int HID = 768;
constexpr int L   = Hh * Ww;        // 1024
constexpr int MR  = Bn * L;         // 8192 rows

// ---- workspace layout (bytes, all 256-aligned) ----
constexpr size_t OFF_HX    = 0;                           // f16 [8192,192]  (LN1 out, reused for LN2 out)
constexpr size_t OFF_WBUF  = OFF_HX    + 3145728;         // f16 weight staging (max 147456 el)
constexpr size_t OFF_XZ    = OFF_WBUF  + 294912;          // f32 [8192,768]  (xz; reused as fc1 out)
constexpr size_t OFF_U16   = OFF_XZ    + 25165824;        // f16 [8192,384]  (u, zigzag order)
constexpr size_t OFF_XDBL  = OFF_U16   + 6291456;         // f32 [8192,48]   (dt|B|C padded)
constexpr size_t OFF_DTP   = OFF_XDBL  + 1572864;         // f16 [8192,32]   (dt padded)
constexpr size_t OFF_DELTA = OFF_DTP   + 524288;          // f32 [8192,384]  (raw delta; reused as gelu f16 [8192,768])
constexpr size_t OFF_YZ    = OFF_DELTA + 12582912;        // f32 [8192,384]  (scan out; reused as proj tmp [8192,192])
constexpr size_t OFF_YG    = OFF_YZ    + 12582912;        // f16 [8192,384]  (gated/normed y)

// ===================================================================
// f32 [n,k] -> f16 [np,kp] with zero padding
// ===================================================================
__global__ void cvt_pad_f16_kernel(const float* __restrict__ src, f16* __restrict__ dst,
                                   int n, int k, int np, int kp) {
  int idx = blockIdx.x * blockDim.x + threadIdx.x;
  if (idx >= np * kp) return;
  int i = idx / kp, j = idx % kp;
  dst[idx] = (i < n && j < k) ? (f16)src[i * k + j] : (f16)0.0f;
}

// ===================================================================
// LayerNorm rows (one wave32 per row) -> f16
// ===================================================================
__global__ void ln_rows_f16_kernel(const float* __restrict__ x, const float* __restrict__ g,
                                   const float* __restrict__ b, f16* __restrict__ out,
                                   int nrows, int C) {
  int row  = blockIdx.x * (blockDim.x >> 5) + (threadIdx.x >> 5);
  int lane = threadIdx.x & 31;
  if (row >= nrows) return;
  const float* xr = x + (size_t)row * C;
  float s = 0.f, s2 = 0.f;
  for (int j = lane; j < C; j += 32) { float v = xr[j]; s += v; s2 += v * v; }
  for (int o = 16; o > 0; o >>= 1) { s += __shfl_xor(s, o, 32); s2 += __shfl_xor(s2, o, 32); }
  float mu  = s / (float)C;
  float inv = rsqrtf(s2 / (float)C - mu * mu + 1e-5f);
  f16* orow = out + (size_t)row * C;
  for (int j = lane; j < C; j += 32)
    orow[j] = (f16)((xr[j] - mu) * inv * g[j] + b[j]);
}

// ===================================================================
// WMMA GEMM, register-blocked: each wave computes a (TM*16) x (TN*16)
// macro-tile of D[M,N] = A[M,K] (f16 row-major) * W[N,K]^T (f16 row-major).
// K stepped by 32 (v_wmma_f32_16x16x32_f16); A fragments reused TN times,
// B fragments reused TM times -> ~21 FLOP/B at TM=2,TN=4.
// ===================================================================
union AV16 { v16h v; v8h h[2]; };

template <int TM, int TN>
__global__ void wmma_gemm_tiled_kernel(const f16* __restrict__ A, const f16* __restrict__ W,
                                       float* __restrict__ D, int Mm, int Nn, int Kk) {
  int wave  = blockIdx.x * (blockDim.x >> 5) + (threadIdx.x >> 5);
  int ntn   = Nn / (16 * TN);
  int ntm   = Mm / (16 * TM);
  if (wave >= ntm * ntn) return;
  int m0 = (wave / ntn) * (16 * TM);
  int n0 = (wave % ntn) * (16 * TN);
  int lane = threadIdx.x & 31;
  int half = lane >> 4;       // 0: lanes 0-15, 1: lanes 16-31
  int l16  = lane & 15;

  const f16* aBase[TM];
#pragma unroll
  for (int mi = 0; mi < TM; mi++)
    aBase[mi] = A + (size_t)(m0 + mi * 16 + l16) * Kk + (half ? 16 : 0);
  const f16* bBase[TN];
#pragma unroll
  for (int ni = 0; ni < TN; ni++)
    bBase[ni] = W + (size_t)(n0 + ni * 16 + l16) * Kk + (half ? 16 : 0);

  v8f acc[TM][TN];
#pragma unroll
  for (int mi = 0; mi < TM; mi++)
#pragma unroll
    for (int ni = 0; ni < TN; ni++) acc[mi][ni] = (v8f){};

  for (int kb = 0; kb < Kk; kb += 32) {
    AV16 a[TM], b[TN];
#pragma unroll
    for (int mi = 0; mi < TM; mi++) {
      __builtin_prefetch(aBase[mi] + kb + 128, 0, 0);    // global_prefetch_b8
      a[mi].h[0] = *(const v8h*)(aBase[mi] + kb);        // K {0..7} / {16..23}
      a[mi].h[1] = *(const v8h*)(aBase[mi] + kb + 8);    // K {8..15} / {24..31}
    }
#pragma unroll
    for (int ni = 0; ni < TN; ni++) {
      __builtin_prefetch(bBase[ni] + kb + 128, 0, 0);
      b[ni].v = *(const v16h*)(bBase[ni] + kb);          // K {0..15} / {16..31} of column l16
    }
#pragma unroll
    for (int mi = 0; mi < TM; mi++)
#pragma unroll
      for (int ni = 0; ni < TN; ni++)
        acc[mi][ni] = __builtin_amdgcn_wmma_f32_16x16x32_f16(false, a[mi].v, false, b[ni].v,
                                                             (short)0, acc[mi][ni], false, false);
  }

  // C/D layout: VGPR r -> M = 8*half + r, N = l16
#pragma unroll
  for (int mi = 0; mi < TM; mi++)
#pragma unroll
    for (int ni = 0; ni < TN; ni++) {
      float* drow = D + (size_t)(m0 + mi * 16 + half * 8) * Nn + n0 + ni * 16 + l16;
#pragma unroll
      for (int r = 0; r < 8; r++) drow[(size_t)r * Nn] = acc[mi][ni][r];
    }
}

// ===================================================================
// depthwise 3x3 conv (SAME) + bias + SiLU + zigzag permute -> u16 [B*L, DI]
// xz layout: f32 [B*H*W, 768]; xm channel d = col d
// ===================================================================
__global__ void conv_silu_perm_kernel(const float* __restrict__ xz, const float* __restrict__ cw,
                                      const float* __restrict__ cb, const int* __restrict__ perm_rev,
                                      f16* __restrict__ u16) {
  int idx = blockIdx.x * blockDim.x + threadIdx.x;
  if (idx >= Bn * L * DI) return;
  int d = idx % DI;
  int t = idx / DI;
  int p = t % L;
  int b = t / L;
  int i = p >> 5, j = p & 31;
  float acc = cb[d];
  const float* wd = cw + d * 9;
#pragma unroll
  for (int di = -1; di <= 1; di++)
#pragma unroll
    for (int dj = -1; dj <= 1; dj++) {
      int ii = i + di, jj = j + dj;
      if (ii < 0 || ii >= Hh || jj < 0 || jj >= Ww) continue;
      acc += wd[(di + 1) * 3 + (dj + 1)] * xz[(size_t)((b * Hh + ii) * Ww + jj) * (2 * DI) + d];
    }
  float sv = acc / (1.f + expf(-acc));  // silu
  int l = perm_rev[p];                  // zigzag position of spatial p
  u16[(size_t)(b * L + l) * DI + d] = (f16)sv;
}

// ===================================================================
// extract dt (cols 0..11 of xdbl [M,48]) -> f16 [M,32] zero-padded
// ===================================================================
__global__ void dt_extract_kernel(const float* __restrict__ xdbl, f16* __restrict__ dtp) {
  int idx = blockIdx.x * blockDim.x + threadIdx.x;
  if (idx >= MR * 32) return;
  int row = idx >> 5, col = idx & 31;
  dtp[idx] = (col < Rr) ? (f16)xdbl[(size_t)row * 48 + col] : (f16)0.0f;
}

// ===================================================================
// selective scan: one thread per (b,d); state h[16] + A[16] in registers
// ===================================================================
__global__ void scan_kernel(const float* __restrict__ delta_raw, const float* __restrict__ dt_b,
                            const f16* __restrict__ u16, const float* __restrict__ xdbl,
                            const float* __restrict__ A_log, const float* __restrict__ Dp,
                            float* __restrict__ yz) {
  int idx = blockIdx.x * blockDim.x + threadIdx.x;
  if (idx >= Bn * DI) return;
  int d = idx % DI, b = idx / DI;
  float An[S], hs[S];
#pragma unroll
  for (int s = 0; s < S; s++) { An[s] = -expf(A_log[d * S + s]); hs[s] = 0.f; }
  float db = dt_b[d], Dd = Dp[d];
  for (int l = 0; l < L; l++) {
    size_t row = (size_t)b * L + l;
    float draw  = delta_raw[row * DI + d] + db;
    float delta = (draw > 20.f) ? draw : log1pf(expf(draw));   // softplus
    float uval  = (float)u16[row * DI + d];
    float du    = delta * uval;
    const float* bc = xdbl + row * 48;
    float y = 0.f;
#pragma unroll
    for (int s = 0; s < S; s++) {
      float dA = expf(delta * An[s]);
      hs[s] = dA * hs[s] + du * bc[12 + s];   // B at cols 12..27
      y += hs[s] * bc[28 + s];                // C at cols 28..43
    }
    yz[row * DI + d] = y + uval * Dd;
  }
}

// ===================================================================
// un-permute + LayerNorm(DI) + silu(z) gate -> f16 [M, DI] (spatial order)
// ===================================================================
__global__ void ygate_ln_kernel(const float* __restrict__ yz, const float* __restrict__ xz,
                                const int* __restrict__ perm_rev, const float* __restrict__ g,
                                const float* __restrict__ bb, f16* __restrict__ out) {
  int r = blockIdx.x * (blockDim.x >> 5) + (threadIdx.x >> 5);  // spatial row
  if (r >= MR) return;
  int lane = threadIdx.x & 31;
  int b = r >> 10, p = r & 1023;
  const float* yr = yz + ((size_t)b * L + perm_rev[p]) * DI;
  float s = 0.f, s2 = 0.f;
  for (int j = lane; j < DI; j += 32) { float v = yr[j]; s += v; s2 += v * v; }
  for (int o = 16; o > 0; o >>= 1) { s += __shfl_xor(s, o, 32); s2 += __shfl_xor(s2, o, 32); }
  float mu  = s / (float)DI;
  float inv = rsqrtf(s2 / (float)DI - mu * mu + 1e-5f);
  const float* zr = xz + (size_t)r * (2 * DI) + DI;
  f16* orow = out + (size_t)r * DI;
  for (int j = lane; j < DI; j += 32) {
    float v = (yr[j] - mu) * inv * g[j] + bb[j];
    float z = zr[j];
    orow[j] = (f16)(v * (z / (1.f + expf(-z))));
  }
}

// ===================================================================
// elementwise epilogues
// ===================================================================
__global__ void residual_add_kernel(const float* __restrict__ x, const float* __restrict__ t,
                                    float* __restrict__ out, int n) {
  int i = blockIdx.x * blockDim.x + threadIdx.x;
  if (i < n) out[i] = x[i] + t[i];
}

__global__ void bias_gelu_f16_kernel(const float* __restrict__ in, const float* __restrict__ bias,
                                     f16* __restrict__ out, int n, int ncol) {
  int i = blockIdx.x * blockDim.x + threadIdx.x;
  if (i >= n) return;
  float v = in[i] + bias[i % ncol];
  float t = 0.7978845608028654f * (v + 0.044715f * v * v * v);   // tanh-approx GELU (jax default)
  out[i] = (f16)(0.5f * v * (1.f + tanhf(t)));
}

__global__ void final_add_kernel(float* __restrict__ out, const float* __restrict__ t,
                                 const float* __restrict__ bias, int n, int ncol) {
  int i = blockIdx.x * blockDim.x + threadIdx.x;
  if (i < n) out[i] = out[i] + t[i] + bias[i % ncol];
}

// ===================================================================
static inline int nblk(long long n, int t) { return (int)((n + t - 1) / t); }

static inline void launch_gemm(const f16* A, const f16* W, float* D,
                               int Mm, int Nn, int Kk, hipStream_t s) {
  if (Nn % 64 == 0) {
    int waves  = (Mm / 32) * (Nn / 64);
    int blocks = (waves + 7) / 8;        // 8 waves (256 threads) per block
    wmma_gemm_tiled_kernel<2, 4><<<blocks, 256, 0, s>>>(A, W, D, Mm, Nn, Kk);
  } else {                               // N == 48 (x_proj)
    int waves  = (Mm / 32) * (Nn / 48);
    int blocks = (waves + 7) / 8;
    wmma_gemm_tiled_kernel<2, 3><<<blocks, 256, 0, s>>>(A, W, D, Mm, Nn, Kk);
  }
}

extern "C" void kernel_launch(void* const* d_in, const int* in_sizes, int n_in,
                              void* d_out, int out_size, void* d_ws, size_t ws_size,
                              hipStream_t stream) {
  const float* x         = (const float*)d_in[0];
  const int*   perm_rev  = (const int*)  d_in[2];
  const float* ln1_g     = (const float*)d_in[3];
  const float* ln1_b     = (const float*)d_in[4];
  const float* in_proj_w = (const float*)d_in[5];
  const float* conv_w    = (const float*)d_in[6];
  const float* conv_b    = (const float*)d_in[7];
  const float* x_proj_w  = (const float*)d_in[8];
  const float* dt_proj_w = (const float*)d_in[9];
  const float* dt_proj_b = (const float*)d_in[10];
  const float* A_log     = (const float*)d_in[11];
  const float* Dp        = (const float*)d_in[12];
  const float* onorm_g   = (const float*)d_in[13];
  const float* onorm_b   = (const float*)d_in[14];
  const float* out_proj_w= (const float*)d_in[15];
  const float* ln2_g     = (const float*)d_in[16];
  const float* ln2_b     = (const float*)d_in[17];
  const float* fc1_w     = (const float*)d_in[18];
  const float* fc1_b     = (const float*)d_in[19];
  const float* fc2_w     = (const float*)d_in[20];
  const float* fc2_b     = (const float*)d_in[21];

  char* ws = (char*)d_ws;
  f16*   hx16  = (f16*)  (ws + OFF_HX);
  f16*   wbuf  = (f16*)  (ws + OFF_WBUF);
  float* xz    = (float*)(ws + OFF_XZ);     // also fc1 output
  f16*   u16   = (f16*)  (ws + OFF_U16);
  float* xdbl  = (float*)(ws + OFF_XDBL);
  f16*   dtp   = (f16*)  (ws + OFF_DTP);
  float* delta = (float*)(ws + OFF_DELTA);
  f16*   m16   = (f16*)  (ws + OFF_DELTA);  // reuse after scan consumes delta
  float* tmpP  = (float*)(ws + OFF_YZ);     // proj temporaries reuse yz region
  float* yz    = (float*)(ws + OFF_YZ);
  f16*   yg16  = (f16*)  (ws + OFF_YG);
  float* outp  = (float*)d_out;

  // ---- SS2D branch ----
  // 1. LN1 -> f16
  ln_rows_f16_kernel<<<MR / 8, 256, 0, stream>>>(x, ln1_g, ln1_b, hx16, MR, Cc);
  // 2-3. in_proj: [8192,192] x [768,192]^T -> xz [8192,768]
  cvt_pad_f16_kernel<<<nblk(768 * 192, 256), 256, 0, stream>>>(in_proj_w, wbuf, 768, 192, 768, 192);
  launch_gemm(hx16, wbuf, xz, MR, 2 * DI, Cc, stream);
  // 4. depthwise conv + silu + zigzag -> u16
  conv_silu_perm_kernel<<<nblk((long long)Bn * L * DI, 256), 256, 0, stream>>>(xz, conv_w, conv_b, perm_rev, u16);
  // 5-6. x_proj: [8192,384] x [48(pad44),384]^T -> xdbl [8192,48]
  cvt_pad_f16_kernel<<<nblk(48 * 384, 256), 256, 0, stream>>>(x_proj_w, wbuf, Rr + 2 * S, DI, 48, DI);
  launch_gemm(u16, wbuf, xdbl, MR, 48, DI, stream);
  // 7-9. dt_proj: pad dt to K=32; [8192,32] x [384,32(pad12)]^T -> delta [8192,384]
  dt_extract_kernel<<<nblk(MR * 32, 256), 256, 0, stream>>>(xdbl, dtp);
  cvt_pad_f16_kernel<<<nblk(384 * 32, 256), 256, 0, stream>>>(dt_proj_w, wbuf, DI, Rr, DI, 32);
  launch_gemm(dtp, wbuf, delta, MR, DI, 32, stream);
  // 10. selective scan (sequential over L, state in registers)
  scan_kernel<<<nblk(Bn * DI, 256), 256, 0, stream>>>(delta, dt_proj_b, u16, xdbl, A_log, Dp, yz);
  // 11. un-permute + out_norm + silu(z) gate -> f16
  ygate_ln_kernel<<<MR / 8, 256, 0, stream>>>(yz, xz, perm_rev, onorm_g, onorm_b, yg16);
  // 12-13. out_proj: [8192,384] x [192,384]^T -> tmpP [8192,192]
  cvt_pad_f16_kernel<<<nblk(192 * 384, 256), 256, 0, stream>>>(out_proj_w, wbuf, Cc, DI, Cc, DI);
  launch_gemm(yg16, wbuf, tmpP, MR, Cc, DI, stream);
  // 14. residual: d_out = x + out_proj(y)
  residual_add_kernel<<<nblk(MR * Cc, 256), 256, 0, stream>>>(x, tmpP, outp, MR * Cc);

  // ---- MLP branch ----
  // 15. LN2 -> f16 (reuse hx16)
  ln_rows_f16_kernel<<<MR / 8, 256, 0, stream>>>(outp, ln2_g, ln2_b, hx16, MR, Cc);
  // 16-17. fc1: [8192,192] x [768,192]^T -> xz region [8192,768]
  cvt_pad_f16_kernel<<<nblk(768 * 192, 256), 256, 0, stream>>>(fc1_w, wbuf, HID, Cc, HID, Cc);
  launch_gemm(hx16, wbuf, xz, MR, HID, Cc, stream);
  // 18. bias + gelu -> f16 (reuse delta region)
  bias_gelu_f16_kernel<<<nblk((long long)MR * HID, 256), 256, 0, stream>>>(xz, fc1_b, m16, MR * HID, HID);
  // 19-20. fc2: [8192,768] x [192,768]^T -> tmpP [8192,192]
  cvt_pad_f16_kernel<<<nblk(192 * 768, 256), 256, 0, stream>>>(fc2_w, wbuf, Cc, HID, Cc, HID);
  launch_gemm(m16, wbuf, tmpP, MR, Cc, HID, stream);
  // 21. final residual: d_out += fc2(m) + fc2_b
  final_add_kernel<<<nblk(MR * Cc, 256), 256, 0, stream>>>(outp, tmpP, fc2_b, MR * Cc, Cc);
}
